// TF_GAM_52793738002611
// MI455X (gfx1250) — matrix operand order repaired and checked by reading, not compiled, not run on hardware
//
#include <hip/hip_runtime.h>
#include <math.h>

typedef _Float16 v8h  __attribute__((ext_vector_type(8)));
typedef _Float16 v16h __attribute__((ext_vector_type(16)));
typedef float    v8f  __attribute__((ext_vector_type(8)));

constexpr int   B      = 4;
constexpr int   N      = 4096;
constexpr int   D      = 512;
constexpr int   TOPK   = 8;
constexpr float LAMBDA = 0.8f;
constexpr float EPSN   = 1e-12f;

// ---------------------------------------------------------------------------
// Kernel 1: per-row L2 normalize; emit fp32 copy (for the fp32 output math)
// and f16 copy (for WMMA). One wave per row, wave32 shuffle reduction.
// ---------------------------------------------------------------------------
__global__ __launch_bounds__(256)
void gam_normalize(const float* __restrict__ x,
                   float* __restrict__ fF,
                   _Float16* __restrict__ fH) {
    const int row  = blockIdx.x * 8 + (threadIdx.x >> 5);
    const int lane = threadIdx.x & 31;
    const float* xr = x + (size_t)row * D;

    float v[16];
    float s = 0.f;
#pragma unroll
    for (int i = 0; i < 16; ++i) {
        v[i] = xr[lane + 32 * i];
        s += v[i] * v[i];
    }
#pragma unroll
    for (int m = 16; m >= 1; m >>= 1) s += __shfl_xor(s, m, 32);

    const float inv = 1.f / fmaxf(sqrtf(s), EPSN);
    float*    fr = fF + (size_t)row * D;
    _Float16* hr = fH + (size_t)row * D;
#pragma unroll
    for (int i = 0; i < 16; ++i) {
        const float n = v[i] * inv;
        fr[lane + 32 * i] = n;
        hr[lane + 32 * i] = (_Float16)n;
    }
}

// ---------------------------------------------------------------------------
// Kernel 2: fused scores(WMMA) -> LDS stripe -> top-8 -> softmax -> sparse
// attention -> blend -> renormalize.  One workgroup (8 waves / 256 thr) per
// 16-row stripe; 16x4096 fp32 score stripe lives entirely in LDS (256 KB of
// the WGP's 320 KB).
// ---------------------------------------------------------------------------
__global__ __launch_bounds__(256)
void gam_attend(const float* __restrict__ fF,
                const _Float16* __restrict__ fH,
                float* __restrict__ out) {
    extern __shared__ char smem[];
    float* sS   = (float*)smem;                       // [16][N] score stripe
    float* sCV  = sS + 16 * N;                        // [16][128] cand vals
    int*   sCI  = (int*)(sCV + 16 * 128);             // [16][128] cand idx
    float* sW   = (float*)(sCI + 16 * 128);           // [16][TOPK] weights
    int*   sWI  = (int*)(sW + 16 * TOPK);             // [16][TOPK] indices
    float* sRed = (float*)(sWI + 16 * TOPK);          // [16][16] norm partials

    const int b       = blockIdx.x / (N / 16);
    const int rowbase = (blockIdx.x % (N / 16)) * 16;
    const int tid     = threadIdx.x;
    const int wave    = tid >> 5;
    const int lane    = tid & 31;
    const size_t batchOff = (size_t)b * N * D;

    // ---------------- Phase A: score stripe via v_wmma_f32_16x16x32_f16 ----
    {
        // A fragment (16x32 f16 per K-step): lane l holds row M=l%16,
        // K-octets {0..7,16..23} (lanes 0-15) or {8..15,24..31} (lanes 16-31).
        const int arow = rowbase + (lane & 15);
        const _Float16* aptr = fH + batchOff + (size_t)arow * D;
        const int kah = (lane >> 4) * 8;

        v16h afrag[16];
#pragma unroll
        for (int ks = 0; ks < 16; ++ks) {
            const v8h lo = *(const v8h*)(aptr + ks * 32 + kah);
            const v8h hi = *(const v8h*)(aptr + ks * 32 + kah + 16);
            afrag[ks] = __builtin_shufflevector(lo, hi,
                0, 1, 2, 3, 4, 5, 6, 7, 8, 9, 10, 11, 12, 13, 14, 15);
        }

        // B fragment (32x16 f16): lane l holds column N=l%16 (== f row
        // colbase+l%16), K-octets {0..7,16..23} (lanes 0-15) or
        // {8..15,24..31} (lanes 16-31): two 16B reads per K-step.
        const int kbh = (lane >> 4) * 8;
        for (int tile = 0; tile < 32; ++tile) {
            const int colbase = (wave * 32 + tile) * 16;
            const _Float16* bptr =
                fH + batchOff + (size_t)(colbase + (lane & 15)) * D + kbh;
            v8f c = {};
#pragma unroll
            for (int ks = 0; ks < 16; ++ks) {
                const v8h blo = *(const v8h*)(bptr + ks * 32);
                const v8h bhi = *(const v8h*)(bptr + ks * 32 + 16);
                const v16h bf = __builtin_shufflevector(blo, bhi,
                    0, 1, 2, 3, 4, 5, 6, 7, 8, 9, 10, 11, 12, 13, 14, 15);
                c = __builtin_amdgcn_wmma_f32_16x16x32_f16(
                        false, afrag[ks], false, bf, (short)0, c, false, false);
            }
            // C layout: lanes 0-15 -> M=r,N=lane ; lanes 16-31 -> M=r+8.
            const int col = colbase + (lane & 15);
            const int mb  = (lane >> 4) * 8;
#pragma unroll
            for (int r = 0; r < 8; ++r) sS[(mb + r) * N + col] = c[r];
        }
    }
    __syncthreads();

    // ---------------- Phase B: top-8 per row + softmax ---------------------
    const int g = tid >> 4;   // row within stripe, 0..15
    const int t = tid & 15;   // 16 threads cooperate per row
    {
        float tv[TOPK];
        int   tix[TOPK];
#pragma unroll
        for (int j = 0; j < TOPK; ++j) { tv[j] = -INFINITY; tix[j] = 0; }
        const float* rowS = sS + g * N;
        const int c0 = t * (N / 16);
        for (int c = 0; c < N / 16; ++c) {
            const float v = rowS[c0 + c];
            if (v > tv[TOPK - 1]) {
                tv[TOPK - 1]  = v;
                tix[TOPK - 1] = c0 + c;
#pragma unroll
                for (int j = TOPK - 1; j > 0; --j)
                    if (tv[j] > tv[j - 1]) {
                        const float fv = tv[j]; tv[j] = tv[j - 1]; tv[j - 1] = fv;
                        const int   ii = tix[j]; tix[j] = tix[j - 1]; tix[j - 1] = ii;
                    }
            }
        }
#pragma unroll
        for (int j = 0; j < TOPK; ++j) {
            sCV[g * 128 + t * TOPK + j] = tv[j];
            sCI[g * 128 + t * TOPK + j] = tix[j];
        }
    }
    __syncthreads();

    if (t == 0) {
        float* cv = sCV + g * 128;
        int*   ci = sCI + g * 128;
        float bv[TOPK];
        int   bix[TOPK];
#pragma unroll
        for (int j = 0; j < TOPK; ++j) {
            float best = -INFINITY; int bi = 0;
            for (int c = 0; c < 128; ++c)
                if (cv[c] > best) { best = cv[c]; bi = c; }
            bv[j] = best; bix[j] = ci[bi]; cv[bi] = -INFINITY;
        }
        const float mx = bv[0];           // selection is descending
        float w[TOPK];
        float se = 0.f;
#pragma unroll
        for (int j = 0; j < TOPK; ++j) { w[j] = __expf(bv[j] - mx); se += w[j]; }
        const float r = 1.f / se;
#pragma unroll
        for (int j = 0; j < TOPK; ++j) {
            sW[g * TOPK + j]  = w[j] * r;
            sWI[g * TOPK + j] = bix[j];
        }
    }
    __syncthreads();

    // ---------------- Phase C: sparse attention + blend + renormalize ------
    {
        const int rowg = rowbase + g;
        float w[TOPK];
        const float* frows[TOPK];
#pragma unroll
        for (int j = 0; j < TOPK; ++j) {
            w[j]     = sW[g * TOPK + j];
            frows[j] = fF + batchOff + (size_t)sWI[g * TOPK + j] * D;
        }
        const float* self = fF + batchOff + (size_t)rowg * D;

        float gv[32];
        float ss = 0.f;
#pragma unroll
        for (int i = 0; i < 32; ++i) {
            const int d = t + 16 * i;
            float att = 0.f;
#pragma unroll
            for (int j = 0; j < TOPK; ++j) att += w[j] * frows[j][d];
            const float val = LAMBDA * self[d] + (1.f - LAMBDA) * att;
            gv[i] = val;
            ss += val * val;
        }
        sRed[g * 16 + t] = ss;
        __syncthreads();
        float tot = 0.f;
#pragma unroll
        for (int i = 0; i < 16; ++i) tot += sRed[g * 16 + i];
        const float inv = 1.f / fmaxf(sqrtf(tot), EPSN);

        float* orow = out + batchOff + (size_t)rowg * D;
#pragma unroll
        for (int i = 0; i < 32; ++i) orow[t + 16 * i] = gv[i] * inv;
    }
}

// ---------------------------------------------------------------------------
extern "C" void kernel_launch(void* const* d_in, const int* in_sizes, int n_in,
                              void* d_out, int out_size, void* d_ws, size_t ws_size,
                              hipStream_t stream) {
    (void)in_sizes; (void)n_in; (void)out_size; (void)ws_size;
    const float* feats = (const float*)d_in[0];

    float*    fF = (float*)d_ws;                        // 32 MB fp32 normalized
    _Float16* fH = (_Float16*)(fF + (size_t)B * N * D); // 16 MB f16 normalized
    float*    out = (float*)d_out;

    gam_normalize<<<(B * N) / 8, 256, 0, stream>>>(feats, fF, fH);

    const size_t smem =
        (size_t)16 * N * sizeof(float)                    // score stripe
        + 16 * 128 * (sizeof(float) + sizeof(int))        // candidates
        + 16 * TOPK * (sizeof(float) + sizeof(int))       // weights/indices
        + 16 * 16 * sizeof(float);                        // norm partials
    gam_attend<<<B * (N / 16), 256, smem, stream>>>(fF, fH, out);
}
